// FeatureGenerator_64115271795185
// MI455X (gfx1250) — matrix operand order repaired and checked by reading, not compiled, not run on hardware
//
#include <hip/hip_runtime.h>

typedef __attribute__((ext_vector_type(16))) _Float16 v16h;
typedef __attribute__((ext_vector_type(8)))  _Float16 v8h;
typedef __attribute__((ext_vector_type(8)))  float    v8f;
typedef __attribute__((ext_vector_type(4)))  unsigned int v4u;
typedef __attribute__((ext_vector_type(8)))  int v8i;
typedef __attribute__((ext_vector_type(4)))  int v4i;

#define FG_EPS 1e-5f
#define NPTS 500000
#define NVOX 100000
#define ACTP 264   // padded activation row stride (halves): mult of 8 -> 16B aligned rows

__device__ __forceinline__ v16h fg_join(v8h lo, v8h hi) {
  return __builtin_shufflevector(lo, hi, 0,1,2,3,4,5,6,7,8,9,10,11,12,13,14,15);
}

// B fragment for v_wmma_f32_16x16x32_f16. Weights stored transposed+padded:
// w[n*Kp + k]. Lane l holds column n0+(l&15); lanes 0-15 K={k0..k0+7, k0+16..k0+23},
// lanes 16-31: +8 (two b128 loads).
__device__ __forceinline__ v16h fg_bfrag(const _Float16* w, int Kp, int n0, int k0, int lane) {
  int base = (n0 + (lane & 15)) * Kp + k0 + ((lane & 16) ? 8 : 0);
  const v8h* p = (const v8h*)(w + base);
  return fg_join(p[0], p[2]);
}

// A fragment from row-major 16 x K f16 tile (row stride Kp). Lane l holds row
// l&15; lanes 0-15: K = {k0..k0+7, k0+16..k0+23}; lanes 16-31: +8.
__device__ __forceinline__ v16h fg_afrag(const _Float16* act, int Kp, int k0, int lane) {
  const _Float16* r = act + (lane & 15) * Kp + k0 + ((lane & 16) ? 8 : 0);
  const v8h* p0 = (const v8h*)(r);
  const v8h* p1 = (const v8h*)(r + 16);
  return fg_join(*p0, *p1);
}

__device__ __forceinline__ v8f fg_wmma(v16h a, v16h b, v8f c) {
  return __builtin_amdgcn_wmma_f32_16x16x32_f16(false, a, false, b, (short)0, c, false, false);
}

// Deterministic float atomic max via sign-split int atomics.
__device__ __forceinline__ void fg_atomic_max_f32(float* addr, float v) {
  if (v >= 0.0f) atomicMax((int*)addr, __float_as_int(v));
  else           atomicMin((unsigned int*)addr, __float_as_uint(v));
}

// ---------------------------------------------------------------------------
// prep: fold BN/bias into scale+shift, build transposed padded f16 weights.
// ---------------------------------------------------------------------------
__global__ void fg_prep(
    const float* __restrict__ w1, const float* __restrict__ w2,
    const float* __restrict__ w3, const float* __restrict__ w4,
    const float* __restrict__ w5,
    const float* __restrict__ bg0, const float* __restrict__ bb0,
    const float* __restrict__ bm0, const float* __restrict__ bv0,
    const float* __restrict__ lb1, const float* __restrict__ bg1,
    const float* __restrict__ bb1, const float* __restrict__ bm1, const float* __restrict__ bv1,
    const float* __restrict__ lb2, const float* __restrict__ bg2,
    const float* __restrict__ bb2, const float* __restrict__ bm2, const float* __restrict__ bv2,
    const float* __restrict__ lb3, const float* __restrict__ bg3,
    const float* __restrict__ bb3, const float* __restrict__ bm3, const float* __restrict__ bv3,
    const float* __restrict__ lb4,
    _Float16* __restrict__ w1t, _Float16* __restrict__ w2t,
    _Float16* __restrict__ w3t, _Float16* __restrict__ w4t, _Float16* __restrict__ w5t,
    float* __restrict__ s0, float* __restrict__ c0,
    float* __restrict__ s1, float* __restrict__ c1,
    float* __restrict__ s2, float* __restrict__ c2,
    float* __restrict__ s3, float* __restrict__ c3,
    float* __restrict__ c4)
{
  int tid = blockIdx.x * blockDim.x + threadIdx.x;
  if (tid < 256*264) { int n = tid/264, k = tid%264; w4t[tid] = (_Float16)((k<256) ? w4[k*256+n] : 0.0f); }
  if (tid < 256*136) { int n = tid/136, k = tid%136; w3t[tid] = (_Float16)((k<128) ? w3[k*256+n] : 0.0f); }
  if (tid < 128*72)  { int n = tid/72,  k = tid%72;  w2t[tid] = (_Float16)((k<64)  ? w2[k*128+n] : 0.0f); }
  if (tid < 64*40)   { int n = tid/40,  k = tid%40;  w1t[tid] = (_Float16)((k<9)   ? w1[k*64+n]  : 0.0f); }
  if (tid < 16*264)  { int n = tid/264, k = tid%264; w5t[tid] = (_Float16)((k<256) ? w5[k*16+n]  : 0.0f); }
  if (tid < 16)  { float s = (tid<9) ? bg0[tid]*rsqrtf(bv0[tid]+FG_EPS) : 0.0f;
                   s0[tid]=s; c0[tid]=(tid<9) ? bb0[tid]-bm0[tid]*s : 0.0f; }
  if (tid < 64)  { float s = bg1[tid]*rsqrtf(bv1[tid]+FG_EPS); s1[tid]=s; c1[tid]=lb1[tid]*s + bb1[tid]-bm1[tid]*s; }
  if (tid < 128) { float s = bg2[tid]*rsqrtf(bv2[tid]+FG_EPS); s2[tid]=s; c2[tid]=lb2[tid]*s + bb2[tid]-bm2[tid]*s; }
  if (tid < 256) { float s = bg3[tid]*rsqrtf(bv3[tid]+FG_EPS); s3[tid]=s; c3[tid]=lb3[tid]*s + bb3[tid]-bm3[tid]*s;
                   c4[tid]=lb4[tid]; }
}

__global__ void fg_init(unsigned int* __restrict__ pool_u, int* __restrict__ cmax) {
  size_t i = (size_t)blockIdx.x * blockDim.x + threadIdx.x;
  if (i < (size_t)NVOX * 256) pool_u[i] = 0xFF800000u;   // -inf
  if (i < (size_t)NVOX * 4)   cmax[i]   = (int)0x80000000; // INT_MIN
}

__global__ void fg_coord(const int* __restrict__ coord, const int* __restrict__ ri,
                         int* __restrict__ cmax) {
  int p = blockIdx.x * blockDim.x + threadIdx.x;
  if (p >= NPTS) return;
  int s = ri[p];
  atomicMax(&cmax[s*4+0], 0);            // padded batch-index column
  atomicMax(&cmax[s*4+1], coord[p*3+0]);
  atomicMax(&cmax[s*4+2], coord[p*3+1]);
  atomicMax(&cmax[s*4+3], coord[p*3+2]);
}

// ---------------------------------------------------------------------------
// Fused per-point MLP + segment-max scatter. One 16-point tile per wave.
// LDS (halves): W1t 2560 | W2t 9216 | W3t 34816 | W4t 67584 | W5t 4224 |
//   params 2368 | 8 waves * 4224 act  = 154560 halves = 309120 B (<320KB).
// Weights+params staged with one Tensor Data Mover DMA (tensor_load_to_lds).
// ---------------------------------------------------------------------------
#define LW1   0
#define LW2   2560
#define LW3   11776
#define LW4   46592
#define SPAR  118400                 // params, 1184 floats = 2368 halves
#define ACTB  120768
#define ACTW  (16*ACTP)
#define NWAVE 8
#define SMEMH (ACTB + NWAVE*ACTW)    // 154560 halves = 309120 B
#define STAGE_BYTES 241536           // W1t..W5t + params, contiguous in ws & LDS
#define STAGE_U8    (STAGE_BYTES/8)  // 30192 8-byte units (fits 16-bit tile_dim0)

__global__ void __launch_bounds__(32*NWAVE) fg_mlp(
    const float* __restrict__ feats, const int* __restrict__ ri,
    const _Float16* __restrict__ wall, float* __restrict__ pooled)
{
  __shared__ _Float16 smem[SMEMH];

#if __has_builtin(__builtin_amdgcn_tensor_load_to_lds) && \
    __has_builtin(__builtin_amdgcn_s_wait_tensorcnt)
  if (threadIdx.x < 32) {            // one TDM DMA per block, issued by wave 0
    unsigned long long ga = (unsigned long long)wall;
    // D# group0: count=1 | lds_addr=0 | global_addr | type=2
    v4u g0 = { 1u, 0u, (unsigned)(ga & 0xFFFFFFFFull),
               (unsigned)((ga >> 32) & 0x01FFFFFFull) | (2u << 30) };
    // D# group1: data_size=8B; tensor_dim0=STAGE_U8, tensor_dim1=1;
    //            tile_dim0=STAGE_U8, tile_dim1=1; dim0_stride=STAGE_U8.
    v8i g1 = { (int)(3u << 16),
               (int)((STAGE_U8 & 0xFFFFu) << 16),
               (int)(((unsigned)STAGE_U8 >> 16) | (1u << 16)),
               (int)((unsigned)STAGE_U8 << 16),
               1,
               (int)STAGE_U8,
               (int)((unsigned)STAGE_U8 << 16),
               0 };
    v4i gz = { 0, 0, 0, 0 };
#if defined(__clang_major__) && __clang_major__ >= 23
    v8i gz8 = { 0, 0, 0, 0, 0, 0, 0, 0 };
    __builtin_amdgcn_tensor_load_to_lds(g0, g1, gz, gz, gz8, 0);
#else
    __builtin_amdgcn_tensor_load_to_lds(g0, g1, gz, gz, 0);
#endif
    __builtin_amdgcn_s_wait_tensorcnt(0);
  }
#else
  {
    v8h* dst = (v8h*)smem;
    const v8h* src = (const v8h*)wall;
    for (int i = threadIdx.x; i < STAGE_BYTES/16; i += blockDim.x) dst[i] = src[i];
  }
#endif
  __syncthreads();

  const int wave = threadIdx.x >> 5, lane = threadIdx.x & 31;
  const int tile = blockIdx.x * (blockDim.x >> 5) + wave;
  if (tile >= NPTS/16) return;             // wave-uniform: EXEC all-1s at WMMAs
  _Float16* act = smem + ACTB + wave * ACTW;
  const float* sp  = (const float*)(smem + SPAR);
  const float* s0a = sp;        const float* c0a = sp + 16;
  const float* s1a = sp + 32;   const float* c1a = sp + 96;
  const float* s2a = sp + 160;  const float* c2a = sp + 288;
  const float* s3a = sp + 416;  const float* c3a = sp + 672;
  const float* c4a = sp + 928;
  const int r0 = (lane & 16) ? 8 : 0;
  const int cl = lane & 15;

  // ---- L1: A = bn0(feats) 16x32 (K=9 zero-padded), B = W1t, N=64 ----
  v16h a1;
  #pragma unroll
  for (int i = 0; i < 16; ++i) a1[i] = (_Float16)0.0f;
  {
    const float* fr = feats + (size_t)(tile*16 + cl) * 9;
    if (lane < 16) {
      #pragma unroll
      for (int i = 0; i < 8; ++i) a1[i] = (_Float16)(fr[i]*s0a[i] + c0a[i]);
    } else {
      a1[0] = (_Float16)(fr[8]*s0a[8] + c0a[8]);
    }
  }
  #pragma unroll
  for (int nt = 0; nt < 4; ++nt) {
    v8f acc = {0.f,0.f,0.f,0.f,0.f,0.f,0.f,0.f};
    acc = fg_wmma(a1, fg_bfrag(smem + LW1, 40, nt*16, 0, lane), acc);
    int c = nt*16 + cl; float s = s1a[c], t = c1a[c];
    #pragma unroll
    for (int r = 0; r < 8; ++r)
      act[(r0+r)*ACTP + c] = (_Float16)fmaxf(fmaf(acc[r], s, t), 0.0f);
  }

  // ---- L2: K=64 (2 steps), N=128; B in LDS ----
  v16h a2[2];
  #pragma unroll
  for (int k = 0; k < 2; ++k) a2[k] = fg_afrag(act, ACTP, k*32, lane);
  #pragma unroll
  for (int nt = 0; nt < 8; ++nt) {
    v8f acc = {0.f,0.f,0.f,0.f,0.f,0.f,0.f,0.f};
    #pragma unroll
    for (int k = 0; k < 2; ++k)
      acc = fg_wmma(a2[k], fg_bfrag(smem + LW2, 72, nt*16, k*32, lane), acc);
    int c = nt*16 + cl; float s = s2a[c], t = c2a[c];
    #pragma unroll
    for (int r = 0; r < 8; ++r)
      act[(r0+r)*ACTP + c] = (_Float16)fmaxf(fmaf(acc[r], s, t), 0.0f);
  }

  // ---- L3: K=128 (4 steps), N=256; B in LDS ----
  v16h a3[4];
  #pragma unroll
  for (int k = 0; k < 4; ++k) a3[k] = fg_afrag(act, ACTP, k*32, lane);
  #pragma unroll
  for (int nt = 0; nt < 16; ++nt) {
    v8f acc = {0.f,0.f,0.f,0.f,0.f,0.f,0.f,0.f};
    #pragma unroll
    for (int k = 0; k < 4; ++k)
      acc = fg_wmma(a3[k], fg_bfrag(smem + LW3, 136, nt*16, k*32, lane), acc);
    int c = nt*16 + cl; float s = s3a[c], t = c3a[c];
    #pragma unroll
    for (int r = 0; r < 8; ++r)
      act[(r0+r)*ACTP + c] = (_Float16)fmaxf(fmaf(acc[r], s, t), 0.0f);
  }

  // ---- L4: K=256 (8 steps), N=256; B in LDS; bias only,
  //      then segment-max scatter ----
  v16h a4[8];
  #pragma unroll
  for (int k = 0; k < 8; ++k) a4[k] = fg_afrag(act, ACTP, k*32, lane);
  int ri8[8];
  #pragma unroll
  for (int r = 0; r < 8; ++r) ri8[r] = ri[tile*16 + r0 + r];
  #pragma unroll
  for (int nt = 0; nt < 16; ++nt) {
    v8f acc = {0.f,0.f,0.f,0.f,0.f,0.f,0.f,0.f};
    #pragma unroll
    for (int k = 0; k < 8; ++k)
      acc = fg_wmma(a4[k], fg_bfrag(smem + LW4, 264, nt*16, k*32, lane), acc);
    int c = nt*16 + cl; float t = c4a[c];
    #pragma unroll
    for (int r = 0; r < 8; ++r)
      fg_atomic_max_f32(&pooled[(size_t)ri8[r]*256 + c], acc[r] + t);
  }
}

// ---------------------------------------------------------------------------
// Head: relu(pooled @ w5 + b5). One 16-voxel tile per wave, 8 chained WMMAs.
// ---------------------------------------------------------------------------
__global__ void __launch_bounds__(256) fg_head(
    const float* __restrict__ pooled, const _Float16* __restrict__ w5t,
    const float* __restrict__ b5, float* __restrict__ outf)
{
  const int lane = threadIdx.x & 31;
  const int tile = blockIdx.x * (blockDim.x >> 5) + (threadIdx.x >> 5);
  if (tile >= NVOX/16) return;
  const int r0 = (lane & 16) ? 8 : 0;
  const int cl = lane & 15;
  const float* prow = pooled + (size_t)(tile*16 + cl) * 256;
  v8f acc = {0.f,0.f,0.f,0.f,0.f,0.f,0.f,0.f};
  #pragma unroll
  for (int k = 0; k < 8; ++k) {
    int off = k*32 + ((lane & 16) ? 8 : 0);
    v16h a;
    #pragma unroll
    for (int i = 0; i < 8; ++i) a[i]     = (_Float16)prow[off + i];
    #pragma unroll
    for (int i = 0; i < 8; ++i) a[8 + i] = (_Float16)prow[off + 16 + i];
    acc = fg_wmma(a, fg_bfrag(w5t, 264, 0, k*32, lane), acc);
  }
  float bb = b5[cl];
  #pragma unroll
  for (int r = 0; r < 8; ++r)
    outf[(size_t)(tile*16 + r0 + r)*16 + cl] = fmaxf(acc[r] + bb, 0.0f);
}

__global__ void fg_cout(const int* __restrict__ cmax, long long* __restrict__ outc) {
  int i = blockIdx.x * blockDim.x + threadIdx.x;
  if (i < NVOX*4) outc[i] = (long long)cmax[i];
}

// ---------------------------------------------------------------------------
extern "C" void kernel_launch(void* const* d_in, const int* in_sizes, int n_in,
                              void* d_out, int out_size, void* d_ws, size_t ws_size,
                              hipStream_t stream) {
  const float* feats = (const float*)d_in[0];
  const int*   coord = (const int*)d_in[1];
  const int*   ri    = (const int*)d_in[2];
  // d_in[3] = shuffle: outputs are permutation-invariant -> unused.
  const float* bg0=(const float*)d_in[4],  *bb0=(const float*)d_in[5];
  const float* bm0=(const float*)d_in[6],  *bv0=(const float*)d_in[7];
  const float* w1 =(const float*)d_in[8],  *lb1=(const float*)d_in[9];
  const float* bg1=(const float*)d_in[10], *bb1=(const float*)d_in[11];
  const float* bm1=(const float*)d_in[12], *bv1=(const float*)d_in[13];
  const float* w2 =(const float*)d_in[14], *lb2=(const float*)d_in[15];
  const float* bg2=(const float*)d_in[16], *bb2=(const float*)d_in[17];
  const float* bm2=(const float*)d_in[18], *bv2=(const float*)d_in[19];
  const float* w3 =(const float*)d_in[20], *lb3=(const float*)d_in[21];
  const float* bg3=(const float*)d_in[22], *bb3=(const float*)d_in[23];
  const float* bm3=(const float*)d_in[24], *bv3=(const float*)d_in[25];
  const float* w4 =(const float*)d_in[26], *lb4=(const float*)d_in[27];
  const float* w5 =(const float*)d_in[28], *b5 =(const float*)d_in[29];

  char* ws = (char*)d_ws;
  float* pooled = (float*)ws;                                   // NVOX*256 f32
  int*   cmax   = (int*)(ws + (size_t)NVOX*256*4);              // NVOX*4 i32
  size_t o = (size_t)NVOX*256*4 + (size_t)NVOX*4*4;             // 16B aligned
  _Float16* w1t = (_Float16*)(ws + o); o += 64*40*2;
  _Float16* w2t = (_Float16*)(ws + o); o += 128*72*2;
  _Float16* w3t = (_Float16*)(ws + o); o += 256*136*2;
  _Float16* w4t = (_Float16*)(ws + o); o += 256*264*2;
  _Float16* w5t = (_Float16*)(ws + o); o += 16*264*2;
  // Folded BN params: contiguous 1184-float block (staged to LDS with weights).
  float* s0 = (float*)(ws + o); o += 16*4;
  float* c0 = (float*)(ws + o); o += 16*4;
  float* s1 = (float*)(ws + o); o += 64*4;
  float* c1 = (float*)(ws + o); o += 64*4;
  float* s2 = (float*)(ws + o); o += 128*4;
  float* c2 = (float*)(ws + o); o += 128*4;
  float* s3 = (float*)(ws + o); o += 256*4;
  float* c3 = (float*)(ws + o); o += 256*4;
  float* c4 = (float*)(ws + o); o += 256*4;

  fg_prep<<<264, 256, 0, stream>>>(w1, w2, w3, w4, w5,
      bg0, bb0, bm0, bv0,
      lb1, bg1, bb1, bm1, bv1,
      lb2, bg2, bb2, bm2, bv2,
      lb3, bg3, bb3, bm3, bv3,
      lb4,
      w1t, w2t, w3t, w4t, w5t,
      s0, c0, s1, c1, s2, c2, s3, c3, c4);

  {
    size_t tot = (size_t)NVOX * 256;
    fg_init<<<(int)((tot + 255)/256), 256, 0, stream>>>((unsigned int*)pooled, cmax);
  }
  fg_coord<<<(NPTS + 255)/256, 256, 0, stream>>>(coord, ri, cmax);

  fg_mlp<<<(NPTS/16 + NWAVE-1)/NWAVE, 32*NWAVE, 0, stream>>>(feats, ri,
      w1t, pooled);

  fg_head<<<(NVOX/16 + 7)/8, 256, 0, stream>>>(pooled, w5t, b5, (float*)d_out);

  fg_cout<<<(NVOX*4 + 255)/256, 256, 0, stream>>>(cmax,
      (long long*)((char*)d_out + (size_t)NVOX*16*4));
}
